// GroupedQueryAttention_16939351015577
// MI455X (gfx1250) — matrix operand order repaired and checked by reading, not compiled, not run on hardware
//
#include <hip/hip_runtime.h>
#include <hip/hip_bf16.h>

#define B_  2
#define S_  2048
#define D_  2048
#define H_  32
#define KV_ 8
#define HD_ 64
// N_REP = H_/KV_ = 4

typedef __attribute__((ext_vector_type(16))) _Float16 v16h;
typedef __attribute__((ext_vector_type(8)))  _Float16 v8h;
typedef __attribute__((ext_vector_type(8)))  float    v8f;
typedef unsigned int u32;
typedef u32 __attribute__((ext_vector_type(4))) v4u;
typedef int __attribute__((ext_vector_type(4))) v4i;
typedef int __attribute__((ext_vector_type(8))) v8i;

#if defined(__has_builtin)
#if __has_builtin(__builtin_amdgcn_tensor_load_to_lds)
#define HAS_TDM 1
#endif
#endif
#ifndef HAS_TDM
#define HAS_TDM 0
#endif

__device__ __forceinline__ v8f zero8() {
    v8f z;
#pragma unroll
    for (int i = 0; i < 8; ++i) z[i] = 0.0f;
    return z;
}

__device__ __forceinline__ v8f wmma_f16(v16h a, v16h b, v8f c) {
    // v_wmma_f32_16x16x32_f16
    return __builtin_amdgcn_wmma_f32_16x16x32_f16(
        false, a, false, b, (short)0, c, false, false);
}

// A fragment (16 M x 32 K, f16). base -> (m=0,k=0), stride = elements between M rows.
// Lane l (g=l>>4, m=l&15): halves 0..7 = K 8g+0..7 ; halves 8..15 = K 16+8g+0..7.
__device__ __forceinline__ v16h load_a16x32(const _Float16* base, int stride) {
    const int l = threadIdx.x & 31;
    const int g = l >> 4, m = l & 15;
    const _Float16* p = base + (size_t)m * stride + 8 * g;
    v8h lo = *(const v8h*)(p);
    v8h hi = *(const v8h*)(p + 16);
    v16h r;
#pragma unroll
    for (int i = 0; i < 8; ++i) { r[i] = lo[i]; r[i + 8] = hi[i]; }
    return r;
}

// B fragment (32 K x 16 N, f16) from an N-major buffer: row n, contiguous K.
// Lane l (g=l>>4, n=l&15): halves 0..7 = K 8g+0..7 ; halves 8..15 = K 16+8g+0..7.
__device__ __forceinline__ v16h load_b32x16(const _Float16* base, int stride) {
    const int l = threadIdx.x & 31;
    const int g = l >> 4, n = l & 15;
    const _Float16* p = base + (size_t)n * stride + 8 * g;
    v8h lo = *(const v8h*)(p);
    v8h hi = *(const v8h*)(p + 16);
    v16h r;
#pragma unroll
    for (int i = 0; i < 8; ++i) { r[i] = lo[i]; r[i + 8] = hi[i]; }
    return r;
}

// ------------------------------------------------ Tensor Data Mover (gfx1250)
// 2D f16 tile load: rows x cols (cols contiguous, row stride in halves).
// LDS destination rows padded to 40 halves (pad_interval=16 dw, pad_amount=4 dw)
// -> 80B row stride, 16B-aligned for ds_load_b128, bank-skewed.
#define LDS_ROW 40

#if HAS_TDM
__device__ __forceinline__ void tdm_load_2d(u32 lds_off, const _Float16* gptr,
                                            int rows, int cols, int row_stride) {
    unsigned long long ga = (unsigned long long)(uintptr_t)gptr;
    v4u g0;
    g0[0] = 1u;                                          // count=1 (valid), user mode
    g0[1] = lds_off;                                     // lds_addr (bytes)
    g0[2] = (u32)(ga & 0xffffffffull);                   // global_addr[31:0]
    g0[3] = (u32)((ga >> 32) & 0x1ffffffull) | (2u << 30); // addr[56:32] | type=2
    v8i g1;
    g1[0] = (int)((1u << 16)       // data_size = 1 -> 2 bytes
                | (1u << 20)       // pad_enable
                | (3u << 22)       // pad_interval: code 3 = 16 DWORDs (= 32 halves)
                | (3u << 25));     // pad_amount:   code 3 = 4 DWORDs  (= 8 halves)
    g1[1] = (int)(((u32)cols & 0xffffu) << 16);          // tensor_dim0[15:0] @ bits 63:48
    g1[2] = (int)((((u32)cols >> 16) & 0xffffu) | (((u32)rows & 0xffffu) << 16));
    g1[3] = (int)((((u32)rows >> 16) & 0xffffu) | (((u32)cols & 0xffffu) << 16)); // tile_dim0
    g1[4] = (int)((u32)rows & 0xffffu);                  // tile_dim1 | tile_dim2=0
    g1[5] = (int)(u32)row_stride;                        // tensor_dim0_stride[31:0]
    g1[6] = 0;                                           // stride0[47:32] | stride1 lo
    g1[7] = 0;                                           // stride1 hi
    v4i z4; z4[0] = z4[1] = z4[2] = z4[3] = 0;
#if defined(__clang_major__) && (__clang_major__ >= 23)
    v8i z8;
#pragma unroll
    for (int i = 0; i < 8; ++i) z8[i] = 0;
    __builtin_amdgcn_tensor_load_to_lds(g0, g1, z4, z4, z8, 0);
#else
    __builtin_amdgcn_tensor_load_to_lds(g0, g1, z4, z4, 0);
#endif
}
#endif

// Fallback: cooperative copy (all 128 threads), same padded-row LDS layout.
__device__ __forceinline__ void stage_copy(_Float16* dst, const _Float16* src,
                                           int rows, int K) {
    for (int idx = threadIdx.x; idx < rows * 4; idx += 128) {
        const int r = idx >> 2, c = (idx & 3) * 8;
        *(v8h*)&dst[(size_t)r * LDS_ROW + c] = *(const v8h*)&src[(size_t)r * K + c];
    }
}

// ---------------------------------------------------------------- converts
__global__ void cvt_f32_f16(const float* __restrict__ x, _Float16* __restrict__ y, size_t n) {
    size_t i = (size_t)blockIdx.x * blockDim.x + threadIdx.x;
    size_t stride = (size_t)gridDim.x * blockDim.x;
    for (; i < n; i += stride) y[i] = (_Float16)x[i];
}

// W: [K][N] f32 row-major  ->  Wt: [N][K] f16   (K,N multiples of 32)
__global__ void wtrans(const float* __restrict__ W, _Float16* __restrict__ Wt, int K, int N) {
    __shared__ float tile[32][33];
    const int kb = blockIdx.x * 32, nb = blockIdx.y * 32;
    const int tx = threadIdx.x & 31, ty = threadIdx.x >> 5;
#pragma unroll
    for (int r = ty; r < 32; r += 8)
        tile[r][tx] = W[(size_t)(kb + r) * N + nb + tx];
    __syncthreads();
#pragma unroll
    for (int r = ty; r < 32; r += 8)
        Wt[(size_t)(nb + r) * K + kb + tx] = (_Float16)tile[tx][r];
}

// V f16 [B*S][KV*HD]  ->  Vt f16 [B][KV][HD][S]
__global__ void vtrans(const _Float16* __restrict__ V, _Float16* __restrict__ Vt) {
    __shared__ _Float16 tile[32][33];
    const int b  = blockIdx.z / KV_;
    const int kv = blockIdx.z % KV_;
    const int sb = blockIdx.x * 32, hb = blockIdx.y * 32;
    const int tx = threadIdx.x & 31, ty = threadIdx.x >> 5;
    const _Float16* src = V + (size_t)b * S_ * (KV_ * HD_) + kv * HD_;
#pragma unroll
    for (int r = ty; r < 32; r += 8)
        tile[r][tx] = src[(size_t)(sb + r) * (KV_ * HD_) + hb + tx];
    __syncthreads();
    _Float16* dst = Vt + ((size_t)(b * KV_ + kv) * HD_) * S_;
#pragma unroll
    for (int r = ty; r < 32; r += 8)
        dst[(size_t)(hb + r) * S_ + sb + tx] = tile[tx][r];
}

// RoPE in-place on T: [B*S][nheads][HD] f16, pairs (2i,2i+1), freqs f32 [S][HD/2]
__global__ void rope_kernel(_Float16* __restrict__ T, const float* __restrict__ freqs,
                            int nheads, size_t total) {
    size_t idx = (size_t)blockIdx.x * blockDim.x + threadIdx.x;
    if (idx >= total) return;
    const int i = (int)(idx & 31);
    size_t t = idx >> 5;
    const int hh = (int)(t % nheads);
    const size_t tok = t / nheads;
    const int s = (int)(tok % S_);
    const float f = freqs[(size_t)s * 32 + i];
    const float cs = __cosf(f), sn = __sinf(f);
    _Float16* p = T + (tok * nheads + hh) * HD_ + 2 * i;
    const float tr = (float)p[0], ti = (float)p[1];
    p[0] = (_Float16)(tr * cs - ti * sn);
    p[1] = (_Float16)(tr * sn + ti * cs);
}

// ---------------------------------------------------------------- GEMM
// C[M][N] = A[M][K] * Bt[N][K]^T. blockDim=128 (4 waves).
// Block tile: 32(M) x 256(N); wave tile 32x64. K staged 32-deep in LDS, double
// buffered, filled by the Tensor Data Mover (async) and consumed via ds_load.
template <bool OUT_F32>
__global__ void gemm_wmma(const _Float16* __restrict__ A, const _Float16* __restrict__ Bt,
                          void* __restrict__ C, int M, int N, int K) {
    __shared__ _Float16 shA[2][32][LDS_ROW];    //  5.0 KB
    __shared__ _Float16 shB[2][256][LDS_ROW];   // 40.0 KB

    const int wave = threadIdx.x >> 5;
    const int m0 = blockIdx.x * 32;
    const int nb = blockIdx.y * 256;

    v8f acc[2][4];
#pragma unroll
    for (int i = 0; i < 2; ++i)
#pragma unroll
        for (int j = 0; j < 4; ++j) acc[i][j] = zero8();

    // ---- prologue: stage k-tile 0 into buffer 0
#if HAS_TDM
    if (threadIdx.x < 32) {
        tdm_load_2d((u32)(uintptr_t)&shA[0][0][0], A + (size_t)m0 * K, 32, 32, K);
        tdm_load_2d((u32)(uintptr_t)&shB[0][0][0], Bt + (size_t)nb * K, 256, 32, K);
        __builtin_amdgcn_s_wait_tensorcnt(0);
    }
#else
    stage_copy(&shA[0][0][0], A + (size_t)m0 * K, 32, K);
    stage_copy(&shB[0][0][0], Bt + (size_t)nb * K, 256, K);
#endif
    __syncthreads();

    int buf = 0;
    for (int k0 = 0; k0 < K; k0 += 32) {
        // ---- kick off async DMA of next k-tile into the other buffer
        if (k0 + 32 < K) {
#if HAS_TDM
            if (threadIdx.x < 32) {
                tdm_load_2d((u32)(uintptr_t)&shA[buf ^ 1][0][0],
                            A + (size_t)m0 * K + k0 + 32, 32, 32, K);
                tdm_load_2d((u32)(uintptr_t)&shB[buf ^ 1][0][0],
                            Bt + (size_t)nb * K + k0 + 32, 256, 32, K);
            }
#else
            stage_copy(&shA[buf ^ 1][0][0], A + (size_t)m0 * K + k0 + 32, 32, K);
            stage_copy(&shB[buf ^ 1][0][0], Bt + (size_t)nb * K + k0 + 32, 256, K);
#endif
        }
        // ---- compute on current buffer (ds_load fragments -> wmma)
        v16h a0 = load_a16x32(&shA[buf][0][0],  LDS_ROW);
        v16h a1 = load_a16x32(&shA[buf][16][0], LDS_ROW);
#pragma unroll
        for (int j = 0; j < 4; ++j) {
            v16h b = load_b32x16(&shB[buf][wave * 64 + 16 * j][0], LDS_ROW);
            acc[0][j] = wmma_f16(a0, b, acc[0][j]);
            acc[1][j] = wmma_f16(a1, b, acc[1][j]);
        }
#if HAS_TDM
        if (threadIdx.x < 32) __builtin_amdgcn_s_wait_tensorcnt(0);
#endif
        __syncthreads();
        buf ^= 1;
    }

    const int l = threadIdx.x & 31;
    const int g = l >> 4, n = l & 15;
#pragma unroll
    for (int i = 0; i < 2; ++i)
#pragma unroll
        for (int j = 0; j < 4; ++j)
#pragma unroll
            for (int r = 0; r < 8; ++r) {
                const size_t row = m0 + 16 * i + r + 8 * g;
                const size_t col = nb + wave * 64 + 16 * j + n;
                if (OUT_F32) ((float*)C)[row * N + col] = acc[i][j][r];
                else ((_Float16*)C)[row * N + col] = (_Float16)acc[i][j][r];
            }
}

// ---------------------------------------------------------------- Flash attention
// grid (B*H, S/64), blockDim 128 (4 waves); each wave: 16 query rows, HD=64.
// S^T = K*Q^T so the C-layout of S^T is directly the A-fragment of P for P*V.
// K fragments are register double-buffered across chunks; V fragments for the
// current chunk issue at the top of the iteration, ~50 instructions before use.
// __launch_bounds__(128,1): allow full 256-VGPR budget, no spills.
struct KFrags { v16h f[4]; };   // [tile0,hd0] [tile0,hd1] [tile1,hd0] [tile1,hd1]
struct VFrags { v16h f[4]; };   // 4 hd tiles

__device__ __forceinline__ KFrags load_kfrags(const _Float16* Kb, int kc) {
    KFrags k;
    k.f[0] = load_a16x32(Kb + (size_t)kc * (KV_ * HD_), KV_ * HD_);
    k.f[1] = load_a16x32(Kb + (size_t)kc * (KV_ * HD_) + 32, KV_ * HD_);
    k.f[2] = load_a16x32(Kb + (size_t)(kc + 16) * (KV_ * HD_), KV_ * HD_);
    k.f[3] = load_a16x32(Kb + (size_t)(kc + 16) * (KV_ * HD_) + 32, KV_ * HD_);
    return k;
}
__device__ __forceinline__ VFrags load_vfrags(const _Float16* Vb, int kc) {
    VFrags v;
#pragma unroll
    for (int j = 0; j < 4; ++j)
        v.f[j] = load_b32x16(Vb + (size_t)(16 * j) * S_ + kc, S_);
    return v;
}

__global__ __launch_bounds__(128, 1)
void attn_kernel(const _Float16* __restrict__ Q, const _Float16* __restrict__ Kk,
                 const _Float16* __restrict__ Vt, _Float16* __restrict__ O) {
    const int b = blockIdx.x / H_;
    const int h = blockIdx.x % H_;
    const int kvh = h >> 2;                         // N_REP = 4
    const int wave = threadIdx.x >> 5;
    const int q0 = (blockIdx.y * 4 + wave) * 16;
    const int l = threadIdx.x & 31;
    const int g = l >> 4, ln = l & 15;

    const _Float16* Qb = Q + (((size_t)b * S_ + q0) * H_ + h) * HD_;
    const _Float16* Kb = Kk + ((size_t)b * S_ * KV_ + kvh) * HD_;
    const _Float16* Vb = Vt + ((size_t)(b * KV_ + kvh) * HD_) * S_;

    const v16h qb0 = load_b32x16(Qb + 0,  H_ * HD_);
    const v16h qb1 = load_b32x16(Qb + 32, H_ * HD_);

    v8f acc[4];
#pragma unroll
    for (int j = 0; j < 4; ++j) acc[j] = zero8();
    float mrun = -1e30f, lrun = 0.0f;

    const int nch = (q0 + 16 + 31) >> 5;
    KFrags kf = load_kfrags(Kb, 0);
    for (int c = 0; c < nch; ++c) {
        const int kc = c * 32;
        const int kcn = (c + 1 < nch) ? kc + 32 : kc;
        // ---- issue current chunk's V loads + next chunk's K loads up front
        VFrags vf = load_vfrags(Vb, kc);
        KFrags kfn = load_kfrags(Kb, kcn);

        // ---- scores^T for current chunk (K frags already resident)
        v8f st0 = zero8(), st1 = zero8();
        st0 = wmma_f16(kf.f[0], qb0, st0);
        st0 = wmma_f16(kf.f[1], qb1, st0);
        st1 = wmma_f16(kf.f[2], qb0, st1);
        st1 = wmma_f16(kf.f[3], qb1, st1);

        const int q = q0 + ln;
        float cmax = -1e30f;
#pragma unroll
        for (int r = 0; r < 8; ++r) {
            const int key0 = kc + r + 8 * g;
            const int key1 = key0 + 16;
            const float v0 = (key0 <= q) ? st0[r] * 0.125f : -1e30f;
            const float v1 = (key1 <= q) ? st1[r] * 0.125f : -1e30f;
            st0[r] = v0; st1[r] = v1;
            cmax = fmaxf(cmax, fmaxf(v0, v1));
        }
        cmax = fmaxf(cmax, __shfl_xor(cmax, 16));
        const float mnew = fmaxf(mrun, cmax);
        const float cf = __expf(mrun - mnew);
        float psum = 0.0f;
        v16h pa;
#pragma unroll
        for (int r = 0; r < 8; ++r) {
            const float p0 = __expf(st0[r] - mnew);
            const float p1 = __expf(st1[r] - mnew);
            psum += p0 + p1;
            pa[r] = (_Float16)p0;
            pa[r + 8] = (_Float16)p1;
        }
        psum += __shfl_xor(psum, 16);
        lrun = lrun * cf + psum;
        mrun = mnew;
#pragma unroll
        for (int r = 0; r < 8; ++r) {
            const float cfr = __shfl(cf, 8 * g + r);
            acc[0][r] *= cfr; acc[1][r] *= cfr; acc[2][r] *= cfr; acc[3][r] *= cfr;
        }
#pragma unroll
        for (int j = 0; j < 4; ++j)
            acc[j] = wmma_f16(pa, vf.f[j], acc[j]);

        kf = kfn;
    }
    const float rl = 1.0f / lrun;
#pragma unroll
    for (int r = 0; r < 8; ++r) {
        const float rr = __shfl(rl, 8 * g + r);
        const size_t tok = (size_t)b * S_ + q0 + r + 8 * g;
        const size_t base = tok * (H_ * HD_) + (size_t)h * HD_;
#pragma unroll
        for (int j = 0; j < 4; ++j)
            O[base + 16 * j + ln] = (_Float16)(acc[j][r] * rr);
    }
}

// ---------------------------------------------------------------- launch
extern "C" void kernel_launch(void* const* d_in, const int* in_sizes, int n_in,
                              void* d_out, int out_size, void* d_ws, size_t ws_size,
                              hipStream_t stream) {
    const float* x     = (const float*)d_in[0];
    const float* wq    = (const float*)d_in[1];
    const float* wk    = (const float*)d_in[2];
    const float* wv    = (const float*)d_in[3];
    const float* wo    = (const float*)d_in[4];
    const float* freqs = (const float*)d_in[5];
    // d_in[6] = mask (causal, applied analytically), d_in[7] = start_pos (0)

    const size_t M = (size_t)B_ * S_;
    char* w = (char*)d_ws;
    _Float16* xh   = (_Float16*)w; w += M * D_ * 2;
    _Float16* wqt  = (_Float16*)w; w += (size_t)D_ * (H_ * HD_) * 2;
    _Float16* wkt  = (_Float16*)w; w += (size_t)D_ * (KV_ * HD_) * 2;
    _Float16* wvt  = (_Float16*)w; w += (size_t)D_ * (KV_ * HD_) * 2;
    _Float16* wot  = (_Float16*)w; w += (size_t)(H_ * HD_) * D_ * 2;
    _Float16* Qb   = (_Float16*)w; w += M * (H_ * HD_) * 2;
    _Float16* Kb   = (_Float16*)w; w += M * (KV_ * HD_) * 2;
    _Float16* Vb   = (_Float16*)w; w += M * (KV_ * HD_) * 2;
    _Float16* Vtb  = (_Float16*)w; w += M * (KV_ * HD_) * 2;
    _Float16* attn = (_Float16*)w; w += M * (H_ * HD_) * 2;

    cvt_f32_f16<<<4096, 256, 0, stream>>>(x, xh, M * D_);
    wtrans<<<dim3(D_ / 32, (H_ * HD_) / 32), 256, 0, stream>>>(wq, wqt, D_, H_ * HD_);
    wtrans<<<dim3(D_ / 32, (KV_ * HD_) / 32), 256, 0, stream>>>(wk, wkt, D_, KV_ * HD_);
    wtrans<<<dim3(D_ / 32, (KV_ * HD_) / 32), 256, 0, stream>>>(wv, wvt, D_, KV_ * HD_);
    wtrans<<<dim3((H_ * HD_) / 32, D_ / 32), 256, 0, stream>>>(wo, wot, H_ * HD_, D_);

    gemm_wmma<false><<<dim3(M / 32, (H_ * HD_) / 256), 128, 0, stream>>>(
        xh, wqt, (void*)Qb, (int)M, H_ * HD_, D_);
    gemm_wmma<false><<<dim3(M / 32, (KV_ * HD_) / 256), 128, 0, stream>>>(
        xh, wkt, (void*)Kb, (int)M, KV_ * HD_, D_);
    gemm_wmma<false><<<dim3(M / 32, (KV_ * HD_) / 256), 128, 0, stream>>>(
        xh, wvt, (void*)Vb, (int)M, KV_ * HD_, D_);

    {
        size_t tq = M * H_ * (HD_ / 2);
        size_t tk = M * KV_ * (HD_ / 2);
        rope_kernel<<<(unsigned)((tq + 255) / 256), 256, 0, stream>>>(Qb, freqs, H_, tq);
        rope_kernel<<<(unsigned)((tk + 255) / 256), 256, 0, stream>>>(Kb, freqs, KV_, tk);
    }

    vtrans<<<dim3(S_ / 32, HD_ / 32, B_ * KV_), 256, 0, stream>>>(Vb, Vtb);

    attn_kernel<<<dim3(B_ * H_, S_ / 64), 128, 0, stream>>>(Qb, Kb, Vtb, attn);

    gemm_wmma<true><<<dim3(M / 32, D_ / 256), 128, 0, stream>>>(
        attn, wot, d_out, (int)M, D_, H_ * HD_);
}
